// Llama4Attention_36215164240703
// MI455X (gfx1250) — compile-verified, matrix operand order checked by reading, not run on hardware
//
#include <hip/hip_runtime.h>
#include <hip/hip_bf16.h>
#include <stdint.h>

// ---------------------------------------------------------------------------
// Types / helpers
// ---------------------------------------------------------------------------
typedef unsigned short u16;
typedef unsigned int   u32;
typedef __bf16 bf16_t;
typedef bf16_t bf16x16 __attribute__((ext_vector_type(16)));
typedef float  floatx8 __attribute__((ext_vector_type(8)));
typedef int    v4i     __attribute__((ext_vector_type(4)));

#define AS1 __attribute__((address_space(1)))
#define AS3 __attribute__((address_space(3)))

union FragAB { bf16x16 v; u32 u[8]; };

#if defined(__has_builtin)
#if __has_builtin(__builtin_amdgcn_global_load_async_to_lds_b128)
#define HAVE_ASYNC_LDS 1
#endif
#endif

__device__ __forceinline__ u16 f2bf(float f) {
    union { float f; u32 u; } c; c.f = f;
    u32 r = c.u + 0x7FFFu + ((c.u >> 16) & 1u);   // round-to-nearest-even
    return (u16)(r >> 16);
}
__device__ __forceinline__ float bf2f(u16 h) {
    union { u32 u; float f; } c; c.u = ((u32)h) << 16;
    return c.f;
}

__device__ __forceinline__ floatx8 wmma_bf16(const FragAB& a, const FragAB& b, floatx8 c) {
    return __builtin_amdgcn_wmma_f32_16x16x32_bf16(
        /*neg_a=*/false, a.v, /*neg_b=*/false, b.v,
        /*c_mod=*/(short)0, c, /*reuse_a=*/false, /*reuse_b=*/false);
}

// Problem constants
#define T_LEN   4096
#define HID     1024
#define NH      8
#define NKV     8
#define HS      128
#define NSLOTS  8192
#define QKV_N   3072

// ---------------------------------------------------------------------------
// f32 -> bf16 convert
// ---------------------------------------------------------------------------
__global__ void cvt_f32_bf16(const float* __restrict__ src, u16* __restrict__ dst, int n) {
    int i = blockIdx.x * blockDim.x + threadIdx.x;
    if (i < n) dst[i] = f2bf(src[i]);
}

// f32 copy (cache background fill)
__global__ void copy_f32(const float* __restrict__ src, float* __restrict__ dst, int n) {
    int i = blockIdx.x * blockDim.x + threadIdx.x;
    if (i < n) dst[i] = src[i];
}

// ---------------------------------------------------------------------------
// WMMA GEMM: C[M,N] = A[M,K] * B[N,K]^T (both bf16, K contiguous).
// Block = 256 threads (8 wave32s) computes a 128x64 macro-tile:
//   - B k-slab (64 cols x 32 k, 4 KB) staged in LDS via async copy,
//     double-buffered, shared by all 8 waves (8x B traffic reduction)
//   - each wave owns a 16x64 strip; A-frag reused across 4 WMMAs
// ---------------------------------------------------------------------------
__global__ void wmma_gemm_bf16_nt(const u16* __restrict__ A, const u16* __restrict__ B,
                                  float* __restrict__ Cf, u16* __restrict__ Cbf,
                                  int M, int N, int K) {
    __shared__ u16 sB[2][64 * 32];        // double-buffered B slab (2 x 4 KB)
    const int tid  = threadIdx.x;
    const int lane = tid & 31;
    const int wid  = tid >> 5;
    const int tilesN = N >> 6;
    const int m0 = (blockIdx.x / tilesN) * 128 + wid * 16;
    const int n0 = (blockIdx.x % tilesN) * 64;
    const int g  = lane >> 4;             // lane group (0/1)
    const int mr = lane & 15;             // row (A) / col (B,C) within tile

    // Cooperative async stage of one 64x32 B slab: 256 threads x 16 B each.
    const int sn = tid >> 2;              // 0..63 : B row (output col)
    const int sk = (tid & 3) * 8;         // 0,8,16,24 (u16 units) = 16 B chunks
    auto stage = [&](int k0, int buf) {
        const u16* gp = B + (size_t)(n0 + sn) * K + k0 + sk;
        u16* lp = &sB[buf][sn * 32 + sk];
#if HAVE_ASYNC_LDS
        __builtin_amdgcn_global_load_async_to_lds_b128(
            (AS1 v4i*)(AS1 void*)(u16*)gp,
            (AS3 v4i*)(AS3 void*)lp, 0, 0);
#else
        *(uint4*)lp = *(const uint4*)gp;
#endif
    };

    floatx8 acc[4] = {};
    const int nk = K >> 5;
    stage(0, 0);
    for (int ks = 0; ks < nk; ++ks) {
        const int cur = ks & 1;
        const bool havenext = (ks + 1) < nk;
        if (havenext) stage((ks + 1) << 5, cur ^ 1);
#if HAVE_ASYNC_LDS
        if (havenext) asm volatile("s_wait_asynccnt 1" ::: "memory");
        else          asm volatile("s_wait_asynccnt 0" ::: "memory");
#endif
        __syncthreads();                  // slab `cur` visible to all waves

        const int k0 = ks << 5;
        FragAB a;
        const u16* ap = A + (size_t)(m0 + mr) * K + k0;
#pragma unroll
        for (int v = 0; v < 8; ++v) {
            const int kb = ((v & 4) << 2) + ((v & 3) << 1) + (g << 3);
            a.u[v] = *(const u32*)(ap + kb);
        }
#pragma unroll
        for (int j = 0; j < 4; ++j) {
            FragAB b;
            const u16* bp = &sB[cur][(j * 16 + mr) * 32];
#pragma unroll
            for (int v = 0; v < 8; ++v) {
                const int kb = ((v & 4) << 2) + ((v & 3) << 1) + (g << 3);
                b.u[v] = *(const u32*)(bp + kb);
            }
            acc[j] = wmma_bf16(a, b, acc[j]);
        }
        __syncthreads();                  // all reads done before slab reuse
    }
#pragma unroll
    for (int j = 0; j < 4; ++j) {
#pragma unroll
        for (int v = 0; v < 8; ++v) {
            const int row = m0 + v + (g << 3);
            const int col = n0 + j * 16 + mr;
            const size_t idx = (size_t)row * N + col;
            if (Cf) Cf[idx] = acc[j][v];
            else    Cbf[idx] = f2bf(acc[j][v]);
        }
    }
}

// ---------------------------------------------------------------------------
// l2norm + RoPE + cache scatter. One block (128 threads) per token.
// Writes: qn/kn bf16 (token-major), vt bf16 (d-major: [h][d][t]),
// f32 K/V caches at slots[t].
// ---------------------------------------------------------------------------
__global__ void norm_rope_cache_kernel(const u16* __restrict__ qkv,
                                       const float* __restrict__ cosb,
                                       const float* __restrict__ sinb,
                                       const int* __restrict__ slots,
                                       u16* __restrict__ qn, u16* __restrict__ kn,
                                       u16* __restrict__ vt,
                                       float* __restrict__ cK, float* __restrict__ cV) {
    const int t = blockIdx.x;
    const int d = threadIdx.x;           // 0..127
    __shared__ float red[128];
    __shared__ float buf[128];
    const int slot = slots[t];
    const int i = d >> 1;
    const float cs = cosb[t * 64 + i];
    const float sn = sinb[t * 64 + i];

    for (int h = 0; h < 16; ++h) {       // h<8: Q heads, h>=8: K heads
        const float x = bf2f(qkv[(size_t)t * QKV_N + h * HS + d]);
        red[d] = x * x;
        __syncthreads();
        for (int s = 64; s > 0; s >>= 1) {
            if (d < s) red[d] += red[d + s];
            __syncthreads();
        }
        const float mean = red[0] * (1.0f / 128.0f);
        const float xn = x * rsqrtf(mean + 1e-6f);
        buf[d] = xn;
        __syncthreads();
        const float xr = buf[2 * i], xi = buf[2 * i + 1];
        const float o = (d & 1) ? (xr * sn + xi * cs) : (xr * cs - xi * sn);
        if (h < 8) {
            qn[(size_t)t * HID + h * HS + d] = f2bf(o);
        } else {
            const int hk = h - 8;
            kn[(size_t)t * HID + hk * HS + d] = f2bf(o);
            cK[(size_t)slot * (NKV * HS) + hk * HS + d] = o;
        }
        __syncthreads();                  // protect red/buf reuse
    }
    for (int h = 0; h < NKV; ++h) {
        const float xv = bf2f(qkv[(size_t)t * QKV_N + 2 * HID + h * HS + d]);
        cV[(size_t)slot * (NKV * HS) + h * HS + d] = xv;
        vt[((size_t)h * HS + d) * T_LEN + t] = f2bf(xv);   // d-major for PV B-frags
    }
}

// ---------------------------------------------------------------------------
// Causal flash attention. One wave per (head, 16-query block) -> 16x128 output.
// Online softmax; S via WMMA (QK^T), P re-fragmented through wave-private LDS,
// PV via WMMA with d-major V.
// ---------------------------------------------------------------------------
__global__ void attention_kernel(const u16* __restrict__ qn, const u16* __restrict__ kn,
                                 const u16* __restrict__ vt, u16* __restrict__ attnOut) {
    __shared__ u16 sP[8 * 512];          // 8 waves x (16x32) bf16 tile
    const int lane = threadIdx.x & 31;
    const int wid  = threadIdx.x >> 5;
    const int wg   = blockIdx.x * 8 + wid;
    const int head = wg >> 8;            // 8 heads x 256 q-blocks
    const int q0   = (wg & 255) << 4;
    const int g    = lane >> 4;
    const int mr   = lane & 15;
    u16* myP = sP + wid * 512;

    // Q A-fragments (16 x 128 = 4 x (16x32))
    FragAB qf[4];
#pragma unroll
    for (int c = 0; c < 4; ++c) {
        const u16* qp = qn + (size_t)(q0 + mr) * HID + head * HS + c * 32;
#pragma unroll
        for (int v = 0; v < 8; ++v) {
            const int kb = ((v & 4) << 2) + ((v & 3) << 1) + (g << 3);
            qf[c].u[v] = *(const u32*)(qp + kb);
        }
    }

    floatx8 o[8] = {};
    float mrow[8], lrow[8];
#pragma unroll
    for (int v = 0; v < 8; ++v) { mrow[v] = -1e30f; lrow[v] = 0.0f; }

    const float scale = 0.0883883476483184f;      // 1/sqrt(128)
    const int nkc = (q0 + 16 + 31) >> 5;          // 32-key chunks up to diagonal

    for (int kc = 0; kc < nkc; ++kc) {
        float s[2][8];
#pragma unroll
        for (int tt = 0; tt < 2; ++tt) {
            const int kt = kc * 32 + tt * 16;
            floatx8 sa = {};
#pragma unroll
            for (int c = 0; c < 4; ++c) {
                FragAB kf;                         // B-frag: K^T tile (HS x 16 keys)
                const u16* kp = kn + (size_t)(kt + mr) * HID + head * HS + c * 32;
#pragma unroll
                for (int v = 0; v < 8; ++v) {
                    const int kb = ((v & 4) << 2) + ((v & 3) << 1) + (g << 3);
                    kf.u[v] = *(const u32*)(kp + kb);
                }
                sa = wmma_bf16(qf[c], kf, sa);
            }
#pragma unroll
            for (int v = 0; v < 8; ++v) {
                const int row = q0 + v + (g << 3);
                const int col = kt + mr;
                s[tt][v] = (col <= row) ? sa[v] * scale : -1e30f;
            }
        }
        // Online softmax per row (rows live on fixed (v,g); reduce over 16 lanes)
#pragma unroll
        for (int v = 0; v < 8; ++v) {
            float mx = fmaxf(s[0][v], s[1][v]);
#pragma unroll
            for (int off = 8; off; off >>= 1) mx = fmaxf(mx, __shfl_xor(mx, off, 16));
            const float mnew = fmaxf(mrow[v], mx);
            const float corr = __expf(mrow[v] - mnew);
            const float p0 = __expf(s[0][v] - mnew);
            const float p1 = __expf(s[1][v] - mnew);
            float rs = p0 + p1;
#pragma unroll
            for (int off = 8; off; off >>= 1) rs += __shfl_xor(rs, off, 16);
            lrow[v] = lrow[v] * corr + rs;
            mrow[v] = mnew;
#pragma unroll
            for (int j = 0; j < 8; ++j) o[j][v] *= corr;
            const int rl = v + (g << 3);
            myP[rl * 32 + mr]      = f2bf(p0);     // C-layout -> row-major LDS tile
            myP[rl * 32 + 16 + mr] = f2bf(p1);
        }
        asm volatile("s_wait_dscnt 0" ::: "memory");   // wave-private LDS RAW

        // P as A-fragment (16x32)
        FragAB pf;
#pragma unroll
        for (int v = 0; v < 8; ++v) {
            const int kb = ((v & 4) << 2) + ((v & 3) << 1) + (g << 3);
            pf.u[v] = *(const u32*)(&myP[mr * 32 + kb]);
        }
        // V B-frags: d-major V => key pairs are contiguous
        const u16* vbase = vt + (size_t)head * HS * T_LEN + (size_t)kc * 32;
#pragma unroll
        for (int j = 0; j < 8; ++j) {
            FragAB vf;
            const u16* vp = vbase + (size_t)(j * 16 + mr) * T_LEN;
#pragma unroll
            for (int v = 0; v < 8; ++v) {
                const int kb = ((v & 4) << 2) + ((v & 3) << 1) + (g << 3);
                vf.u[v] = *(const u32*)(vp + kb);
            }
            o[j] = wmma_bf16(pf, vf, o[j]);
        }
    }
    // Epilogue: normalize by row sums, emit bf16 attn (t, NH*HS)
#pragma unroll
    for (int v = 0; v < 8; ++v) {
        const float inv = 1.0f / lrow[v];
        const int row = q0 + v + (g << 3);
#pragma unroll
        for (int j = 0; j < 8; ++j) {
            attnOut[(size_t)row * HID + head * HS + j * 16 + mr] = f2bf(o[j][v] * inv);
        }
    }
}

// ---------------------------------------------------------------------------
// Host-side launch
// ---------------------------------------------------------------------------
extern "C" void kernel_launch(void* const* d_in, const int* in_sizes, int n_in,
                              void* d_out, int out_size, void* d_ws, size_t ws_size,
                              hipStream_t stream) {
    (void)in_sizes; (void)n_in; (void)out_size; (void)ws_size;
    const float* hidden = (const float*)d_in[0];
    const float* cosb   = (const float*)d_in[1];
    const float* sinb   = (const float*)d_in[2];
    const float* w_qkv  = (const float*)d_in[3];
    const float* w_o    = (const float*)d_in[4];
    const int*   slots  = (const int*)  d_in[5];
    const float* cacheK = (const float*)d_in[6];
    const float* cacheV = (const float*)d_in[7];

    float* out   = (float*)d_out;                       // 4096*1024
    float* outCK = out + (size_t)T_LEN * HID;           // 8192*8*128
    float* outCV = outCK + (size_t)NSLOTS * NKV * HS;

    // Workspace layout (bytes)
    char* ws = (char*)d_ws;
    u16* qkv_bf  = (u16*)(ws);                                  // 4096*3072
    u16* h_bf    = (u16*)(ws + 25165824);                       // 4096*1024
    u16* wqkv_bf = (u16*)(ws + 33554432);                       // 3072*1024
    u16* wo_bf   = (u16*)(ws + 39845888);                       // 1024*1024
    u16* qn_bf   = (u16*)(ws + 41943040);                       // 4096*1024
    u16* kn_bf   = (u16*)(ws + 50331648);                       // 4096*1024
    u16* vt_bf   = (u16*)(ws + 58720256);                       // 8*128*4096 (d-major)
    u16* attn_bf = (u16*)(ws + 67108864);                       // 4096*1024

    // 1) f32 -> bf16 converts
    cvt_f32_bf16<<<(T_LEN * HID + 255) / 256, 256, 0, stream>>>(hidden, h_bf, T_LEN * HID);
    cvt_f32_bf16<<<(QKV_N * HID + 255) / 256, 256, 0, stream>>>(w_qkv, wqkv_bf, QKV_N * HID);
    cvt_f32_bf16<<<(HID * HID + 255) / 256, 256, 0, stream>>>(w_o, wo_bf, HID * HID);

    // 2) QKV projection: (4096x1024) x (3072x1024)^T -> bf16 qkv
    {
        const int blocks = (T_LEN / 128) * (QKV_N / 64);  // 32*48 = 1536
        wmma_gemm_bf16_nt<<<blocks, 256, 0, stream>>>(h_bf, wqkv_bf, nullptr, qkv_bf,
                                                      T_LEN, QKV_N, HID);
    }

    // 3) Background cache copy (slots only cover 4096 of 8192)
    {
        const int n = NSLOTS * NKV * HS;                 // 8,388,608
        copy_f32<<<(n + 255) / 256, 256, 0, stream>>>(cacheK, outCK, n);
        copy_f32<<<(n + 255) / 256, 256, 0, stream>>>(cacheV, outCV, n);
    }

    // 4) l2norm + RoPE + cache scatter
    norm_rope_cache_kernel<<<T_LEN, 128, 0, stream>>>(qkv_bf, cosb, sinb, slots,
                                                      qn_bf, kn_bf, vt_bf, outCK, outCV);

    // 5) Causal flash attention (2048 waves)
    attention_kernel<<<256, 256, 0, stream>>>(qn_bf, kn_bf, vt_bf, attn_bf);

    // 6) Output projection: (4096x1024) x (1024x1024)^T -> f32 out
    {
        const int blocks = (T_LEN / 128) * (HID / 64);   // 32*16 = 512
        wmma_gemm_bf16_nt<<<blocks, 256, 0, stream>>>(attn_bf, wo_bf, out, nullptr,
                                                      T_LEN, HID, HID);
    }
}
